// HierarchicalDownTopFusion_3633542333117
// MI455X (gfx1250) — compile-verified, matrix operand order checked by reading, not run on hardware
//
#include <hip/hip_runtime.h>
#include <math.h>

typedef __attribute__((ext_vector_type(16))) __bf16        v16bf;
typedef __attribute__((ext_vector_type(8)))  float         v8f;
typedef __attribute__((ext_vector_type(8)))  unsigned int  v8u;
typedef __attribute__((ext_vector_type(4)))  float         f4;
typedef __attribute__((ext_vector_type(2)))  unsigned int  u2;

#define B_ROWS 65536
#define IN_DIM 768
#define HID    256
#define TILE_M 16

// ---------- helpers ----------

__device__ __forceinline__ unsigned short f2bf(float f) {
  unsigned int x = __builtin_bit_cast(unsigned int, f);
  unsigned int r = x + 0x7FFFu + ((x >> 16) & 1u);   // round-to-nearest-even
  return (unsigned short)(r >> 16);
}

__device__ __forceinline__ float gelu_exact(float x) {
  return 0.5f * x * (1.0f + erff(x * 0.70710678118654752f));
}

__device__ __forceinline__ v8f wmma_bf16(v16bf a, v16bf b, v8f c) {
  return __builtin_amdgcn_wmma_f32_16x16x32_bf16(
      /*neg_a=*/false, a, /*neg_b=*/false, b,
      /*c_mod=*/(short)0, c, /*reuse_a=*/false, /*reuse_b=*/false);
}

// A fragment (16x32 bf16) from row-major bf16 LDS buffer, viewed as u32.
// Per ISA 16-bit A layout: lane 0-15 hold M=lane; dword d holds K pair:
//   k = kk*32 + (lane<16?0:8) + (d>=4?16:0) + 2*(d&3)  (pair k,k+1 in one dword)
// -> d=0..3 and d=4..7 are each 4 contiguous dwords => 2x ds_load_b128.
__device__ __forceinline__ v16bf ldA(const unsigned int* buf, int strideU32,
                                     int kk, int lane) {
  int row  = lane & 15;
  int base = (lane < 16) ? 0 : 4;     // in u32 units (8 bf16)
  v8u u;
#pragma unroll
  for (int d = 0; d < 8; ++d) {
    u[d] = buf[row * strideU32 + kk * 16 + base + ((d >= 4) ? 8 : 0) + (d & 3)];
  }
  return __builtin_bit_cast(v16bf, u);
}

// B fragment from pre-packed weights, lane-major: lane's 8 dwords contiguous
// (32 B) -> 2x global_load_b128 per fragment, wave reads contiguous 1 KB.
__device__ __forceinline__ v16bf ldB(const unsigned int* __restrict__ PW,
                                     int frag, int lane) {
  const v8u* p = (const v8u*)(PW + (size_t)frag * 256 + lane * 8);
  return __builtin_bit_cast(v16bf, *p);
}

// ---------- prologue: pack fp32 weight [K][N] into WMMA B-frag bf16 order ----------
// Fragment f = kk*(N/16)+nn, 256 u32 each; u32 index inside frag = lane*8 + d.
// Value pair in dword d of lane L:  k_local = (L<16?0:8)+(d>=4?16:0)+2*(d&3),
// n = nn*16 + (L&15);  dword = bf16(W[k][n]) | bf16(W[k+1][n])<<16.
__global__ __launch_bounds__(256) void pack_weight_kernel(
    const float* __restrict__ W, unsigned int* __restrict__ PW, int K, int N) {
  int t       = blockIdx.x * blockDim.x + threadIdx.x;
  int nFragsN = N >> 4;
  int frag    = t >> 8;
  int within  = t & 255;
  int lane    = within >> 3;   // lane-major packing
  int d       = within & 7;
  int kk      = frag / nFragsN;
  int nn      = frag - kk * nFragsN;
  int klocal  = ((lane < 16) ? 0 : 8) + ((d >= 4) ? 16 : 0) + 2 * (d & 3);
  int n       = nn * 16 + (lane & 15);
  int k0      = kk * 32 + klocal;
  unsigned int lo = f2bf(W[(size_t)k0 * N + n]);
  unsigned int hi = f2bf(W[(size_t)(k0 + 1) * N + n]);
  PW[t] = lo | (hi << 16);
}

// ---------- fused hierarchical kernel ----------
__global__ __launch_bounds__(256) void fused_hier_kernel(
    const float* __restrict__ gfeat,   // [B,768]
    const float* __restrict__ mfeat,   // [4,B,768]
    const float* __restrict__ sfeat,   // [9,B,768]
    const float* __restrict__ sm_b1, const float* __restrict__ sm_g,
    const float* __restrict__ sm_be, const float* __restrict__ sm_b2,
    const float* __restrict__ mg_b1, const float* __restrict__ mg_g,
    const float* __restrict__ mg_be, const float* __restrict__ mg_b2,
    const unsigned int* __restrict__ PW_sm1,   // packed 768x256
    const unsigned int* __restrict__ PW_sm2,   // packed 256x768
    const unsigned int* __restrict__ PW_mg1,   // packed 768x256
    const unsigned int* __restrict__ PW_mg2,   // packed 256x768
    float* __restrict__ out) {
  __shared__ unsigned short AbufS[TILE_M * IN_DIM];  // 24 KB (bf16 A tile)
  __shared__ float          Hbuf[TILE_M * HID];      // 16 KB (f32 hidden)
  __shared__ unsigned short GbufS[TILE_M * HID];     //  8 KB (bf16 GELU out)
  __shared__ float          Fbuf[TILE_M * IN_DIM];   // 48 KB (f32 staging)
  __shared__ float redS[256];
  __shared__ float redS2[256];
  __shared__ float muS[TILE_M];
  __shared__ float rsS[TILE_M];

  const int tid  = threadIdx.x;
  const int lane = tid & 31;
  const int wv   = tid >> 5;
  const int row0 = blockIdx.x * TILE_M;

  const unsigned int* Abuf_u32 = (const unsigned int*)AbufS;
  const unsigned int* Gbuf_u32 = (const unsigned int*)GbufS;
  unsigned int*       Abuf_w32 = (unsigned int*)AbufS;
  f4*                 Fbuf4    = (f4*)Fbuf;

  const int adj[4][4] = {{0, 1, 3, 4}, {1, 2, 4, 5}, {3, 4, 6, 7}, {4, 5, 7, 8}};

  v8f acc2[6];
  v8f vzero = {};
#pragma unroll
  for (int i = 0; i < 6; ++i) acc2[i] = vzero;

#pragma unroll 1   // keep rolled: prevents cross-iteration weight hoist + spills
  for (int m = 0; m < 4; ++m) {
    // ---- pool 4 small neighbors -> A tile (bf16), float4 loads ----
    const f4* s0 = (const f4*)(sfeat + (size_t)adj[m][0] * B_ROWS * IN_DIM);
    const f4* s1 = (const f4*)(sfeat + (size_t)adj[m][1] * B_ROWS * IN_DIM);
    const f4* s2 = (const f4*)(sfeat + (size_t)adj[m][2] * B_ROWS * IN_DIM);
    const f4* s3 = (const f4*)(sfeat + (size_t)adj[m][3] * B_ROWS * IN_DIM);
#pragma unroll
    for (int i = 0; i < (TILE_M * IN_DIM) / (4 * 256); ++i) {  // 12 iters
      int e4  = tid + i * 256;
      int row = e4 / (IN_DIM / 4);
      int c4  = e4 - row * (IN_DIM / 4);
      size_t off4 = (size_t)(row0 + row) * (IN_DIM / 4) + c4;
      f4 a = s0[off4], b = s1[off4], c = s2[off4], d = s3[off4];
      f4 v;
#pragma unroll
      for (int j = 0; j < 4; ++j) v[j] = 0.25f * (a[j] + b[j] + c[j] + d[j]);
      u2 packed;
      packed[0] = (unsigned int)f2bf(v[0]) | ((unsigned int)f2bf(v[1]) << 16);
      packed[1] = (unsigned int)f2bf(v[2]) | ((unsigned int)f2bf(v[3]) << 16);
      *(u2*)&Abuf_w32[e4 * 2] = packed;
    }
    __syncthreads();

    // ---- GEMM1: [16x768] @ [768x256]; wave wv owns n-tiles {2wv, 2wv+1} ----
    v8f h0 = vzero, h1 = vzero;
#pragma unroll 1   // rolled: hardware-counter pipelined streaming, no spill
    for (int kk = 0; kk < IN_DIM / 32; ++kk) {  // 24
      v16bf a  = ldA(Abuf_u32, IN_DIM / 2, kk, lane);
      v16bf b0 = ldB(PW_sm1, kk * 16 + wv * 2 + 0, lane);
      v16bf b1 = ldB(PW_sm1, kk * 16 + wv * 2 + 1, lane);
      h0 = wmma_bf16(a, b0, h0);
      h1 = wmma_bf16(a, b1, h1);
    }
    {
      int colb  = wv * 32 + (lane & 15);
      float bi0 = sm_b1[colb];
      float bi1 = sm_b1[colb + 16];
      int rbase = (lane < 16) ? 0 : 8;
#pragma unroll
      for (int r = 0; r < 8; ++r) {
        Hbuf[(rbase + r) * HID + colb]      = h0[r] + bi0;
        Hbuf[(rbase + r) * HID + colb + 16] = h1[r] + bi1;
      }
    }
    __syncthreads();

    // ---- LayerNorm stats over 256 hidden dims per row ----
    {
      int row = tid >> 4, ch = tid & 15;
      float s = 0.f, s2 = 0.f;
#pragma unroll
      for (int c = 0; c < 16; ++c) {
        float x = Hbuf[row * HID + ch * 16 + c];
        s += x;
        s2 += x * x;
      }
      redS[tid]  = s;
      redS2[tid] = s2;
    }
    __syncthreads();
    if (tid < TILE_M) {
      float ts = 0.f, ts2 = 0.f;
#pragma unroll
      for (int c = 0; c < 16; ++c) {
        ts += redS[tid * 16 + c];
        ts2 += redS2[tid * 16 + c];
      }
      float mu  = ts * (1.0f / HID);
      float var = ts2 * (1.0f / HID) - mu * mu;
      muS[tid] = mu;
      rsS[tid] = rsqrtf(var + 1e-5f);
    }
    __syncthreads();

    // ---- LN apply + exact GELU -> G tile (bf16) ----
    {
      float gv = sm_g[tid & 255];
      float bv = sm_be[tid & 255];
#pragma unroll
      for (int j = 0; j < TILE_M; ++j) {
        int e   = j * HID + tid;  // row j, col tid
        float x = Hbuf[e];
        x = (x - muS[j]) * rsS[j] * gv + bv;
        GbufS[e] = f2bf(gelu_exact(x));
      }
    }
    __syncthreads();

    // ---- GEMM2: [16x256] @ [256x768] accumulate across m ----
#pragma unroll 1
    for (int kk = 0; kk < HID / 32; ++kk) {  // 8
      v16bf a = ldA(Gbuf_u32, HID / 2, kk, lane);
#pragma unroll
      for (int nt = 0; nt < 6; ++nt) {
        v16bf b = ldB(PW_sm2, kk * 48 + wv * 6 + nt, lane);
        acc2[nt] = wmma_bf16(a, b, acc2[nt]);
      }
    }
  }

  // ---- cooperative vectorized sum of the 4 medium feats -> Fbuf (f32) ----
  {
    const f4* m0 = (const f4*)(mfeat + 0 * (size_t)B_ROWS * IN_DIM);
    const f4* m1 = (const f4*)(mfeat + 1 * (size_t)B_ROWS * IN_DIM);
    const f4* m2 = (const f4*)(mfeat + 2 * (size_t)B_ROWS * IN_DIM);
    const f4* m3 = (const f4*)(mfeat + 3 * (size_t)B_ROWS * IN_DIM);
#pragma unroll
    for (int i = 0; i < (TILE_M * IN_DIM) / (4 * 256); ++i) {  // 12 iters
      int e4  = tid + i * 256;
      int row = e4 / (IN_DIM / 4);
      int c4  = e4 - row * (IN_DIM / 4);
      size_t off4 = (size_t)(row0 + row) * (IN_DIM / 4) + c4;
      f4 a = m0[off4], b = m1[off4], c = m2[off4], d = m3[off4];
      f4 v;
#pragma unroll
      for (int j = 0; j < 4; ++j) v[j] = a[j] + b[j] + c[j] + d[j];
      Fbuf4[e4] = v;
    }
  }
  __syncthreads();

  // ---- medium_pooled = 0.25*(acc2 + msum) + sm_b2 -> A tile (bf16) ----
  {
    int rbase = (lane < 16) ? 0 : 8;
#pragma unroll
    for (int nt = 0; nt < 6; ++nt) {
      int col   = wv * 96 + nt * 16 + (lane & 15);
      float b2v = sm_b2[col];
#pragma unroll
      for (int r = 0; r < 8; ++r) {
        int row   = rbase + r;
        float val = 0.25f * (acc2[nt][r] + Fbuf[row * IN_DIM + col]) + b2v;
        AbufS[row * IN_DIM + col] = f2bf(val);
      }
      acc2[nt] = vzero;  // reuse as GEMM4 accumulator
    }
  }
  __syncthreads();

  // ---- GEMM3: medium_pooled @ mg_w1 ----
  v8f h0 = vzero, h1 = vzero;
#pragma unroll 1
  for (int kk = 0; kk < IN_DIM / 32; ++kk) {
    v16bf a  = ldA(Abuf_u32, IN_DIM / 2, kk, lane);
    v16bf b0 = ldB(PW_mg1, kk * 16 + wv * 2 + 0, lane);
    v16bf b1 = ldB(PW_mg1, kk * 16 + wv * 2 + 1, lane);
    h0 = wmma_bf16(a, b0, h0);
    h1 = wmma_bf16(a, b1, h1);
  }
  {
    int colb  = wv * 32 + (lane & 15);
    float bi0 = mg_b1[colb];
    float bi1 = mg_b1[colb + 16];
    int rbase = (lane < 16) ? 0 : 8;
#pragma unroll
    for (int r = 0; r < 8; ++r) {
      Hbuf[(rbase + r) * HID + colb]      = h0[r] + bi0;
      Hbuf[(rbase + r) * HID + colb + 16] = h1[r] + bi1;
    }
  }
  __syncthreads();

  // ---- LN stats (stage 2) ----
  {
    int row = tid >> 4, ch = tid & 15;
    float s = 0.f, s2 = 0.f;
#pragma unroll
    for (int c = 0; c < 16; ++c) {
      float x = Hbuf[row * HID + ch * 16 + c];
      s += x;
      s2 += x * x;
    }
    redS[tid]  = s;
    redS2[tid] = s2;
  }
  __syncthreads();
  if (tid < TILE_M) {
    float ts = 0.f, ts2 = 0.f;
#pragma unroll
    for (int c = 0; c < 16; ++c) {
      ts += redS[tid * 16 + c];
      ts2 += redS2[tid * 16 + c];
    }
    float mu  = ts * (1.0f / HID);
    float var = ts2 * (1.0f / HID) - mu * mu;
    muS[tid] = mu;
    rsS[tid] = rsqrtf(var + 1e-5f);
  }
  __syncthreads();

  {
    float gv = mg_g[tid & 255];
    float bv = mg_be[tid & 255];
#pragma unroll
    for (int j = 0; j < TILE_M; ++j) {
      int e   = j * HID + tid;
      float x = Hbuf[e];
      x = (x - muS[j]) * rsS[j] * gv + bv;
      GbufS[e] = f2bf(gelu_exact(x));
    }
  }
  __syncthreads();

  // ---- GEMM4: @ mg_w2 ----
#pragma unroll 1
  for (int kk = 0; kk < HID / 32; ++kk) {
    v16bf a = ldA(Gbuf_u32, HID / 2, kk, lane);
#pragma unroll
    for (int nt = 0; nt < 6; ++nt) {
      v16bf b = ldB(PW_mg2, kk * 48 + wv * 6 + nt, lane);
      acc2[nt] = wmma_bf16(a, b, acc2[nt]);
    }
  }

  // ---- stage acc + mg_b2 into Fbuf, then vectorized residual + store ----
  __syncthreads();  // Fbuf reads (medium sum) all complete before overwrite
  {
    int rbase = (lane < 16) ? 0 : 8;
#pragma unroll
    for (int nt = 0; nt < 6; ++nt) {
      int col   = wv * 96 + nt * 16 + (lane & 15);
      float b2v = mg_b2[col];
#pragma unroll
      for (int r = 0; r < 8; ++r) {
        Fbuf[(rbase + r) * IN_DIM + col] = acc2[nt][r] + b2v;
      }
    }
  }
  __syncthreads();
  {
    const f4* g4 = (const f4*)gfeat;
    f4*       o4 = (f4*)out;
#pragma unroll
    for (int i = 0; i < (TILE_M * IN_DIM) / (4 * 256); ++i) {  // 12 iters
      int e4  = tid + i * 256;
      int row = e4 / (IN_DIM / 4);
      int c4  = e4 - row * (IN_DIM / 4);
      size_t off4 = (size_t)(row0 + row) * (IN_DIM / 4) + c4;
      f4 a = Fbuf4[e4];
      f4 b = g4[off4];
      f4 v;
#pragma unroll
      for (int j = 0; j < 4; ++j) v[j] = a[j] + b[j];
      o4[off4] = v;
    }
  }
}

// ---------- launch ----------

extern "C" void kernel_launch(void* const* d_in, const int* in_sizes, int n_in,
                              void* d_out, int out_size, void* d_ws, size_t ws_size,
                              hipStream_t stream) {
  const float* gfeat  = (const float*)d_in[0];
  const float* mfeat  = (const float*)d_in[1];
  const float* sfeat  = (const float*)d_in[2];
  const float* sm_w1  = (const float*)d_in[3];
  const float* sm_b1  = (const float*)d_in[4];
  const float* sm_g   = (const float*)d_in[5];
  const float* sm_be  = (const float*)d_in[6];
  const float* sm_w2  = (const float*)d_in[7];
  const float* sm_b2  = (const float*)d_in[8];
  const float* mg_w1  = (const float*)d_in[9];
  const float* mg_b1  = (const float*)d_in[10];
  const float* mg_g   = (const float*)d_in[11];
  const float* mg_be  = (const float*)d_in[12];
  const float* mg_w2  = (const float*)d_in[13];
  const float* mg_b2  = (const float*)d_in[14];

  // packed weight buffers: 4 matrices x 98304 u32 (bf16 pairs) = 1.5 MB
  const int PW_U32 = (IN_DIM / 32) * (HID / 16) * 256;  // 98304 (same both shapes)
  unsigned int* PW_sm1 = (unsigned int*)d_ws;
  unsigned int* PW_sm2 = PW_sm1 + PW_U32;
  unsigned int* PW_mg1 = PW_sm2 + PW_U32;
  unsigned int* PW_mg2 = PW_mg1 + PW_U32;

  dim3 pblk(256);
  dim3 pgrd(PW_U32 / 256);  // 384
  pack_weight_kernel<<<pgrd, pblk, 0, stream>>>(sm_w1, PW_sm1, IN_DIM, HID);
  pack_weight_kernel<<<pgrd, pblk, 0, stream>>>(sm_w2, PW_sm2, HID, IN_DIM);
  pack_weight_kernel<<<pgrd, pblk, 0, stream>>>(mg_w1, PW_mg1, IN_DIM, HID);
  pack_weight_kernel<<<pgrd, pblk, 0, stream>>>(mg_w2, PW_mg2, HID, IN_DIM);

  dim3 blk(256);
  dim3 grd(B_ROWS / TILE_M);  // 4096
  fused_hier_kernel<<<grd, blk, 0, stream>>>(
      gfeat, mfeat, sfeat,
      sm_b1, sm_g, sm_be, sm_b2,
      mg_b1, mg_g, mg_be, mg_b2,
      PW_sm1, PW_sm2, PW_mg1, PW_mg2,
      (float*)d_out);
}
